// DependencyParser_26448408609316
// MI455X (gfx1250) — matrix operand order repaired and checked by reading, not compiled, not run on hardware
//
#include <hip/hip_runtime.h>
#include <hip/hip_bf16.h>
#include <math.h>

typedef __attribute__((ext_vector_type(16))) _Float16 v16h;
typedef __attribute__((ext_vector_type(8)))  _Float16 v8h;
typedef __attribute__((ext_vector_type(8)))  float    v8f;

#define BATCH 32
#define SEQ   128
#define DH    125          // LSTM hidden per direction
#define G4    500          // 4*DH
#define H2    250
#define HID   100
#define NL    50
#define ROWS  (BATCH*SEQ)  // 4096
#define XGLD  512          // padded xg row stride
#define WELD  256          // padded f16 stride for 250-wide activations

__device__ __forceinline__ float sigf(float x) { return 1.f / (1.f + expf(-x)); }

// ---- WMMA 16-bit operand fetch (CDNA5 ISA 7.12.2), vectorized as b128 ----
// A 16x32: lanes 0-15 rows, per lane K runs {base..base+7} and {base+16..base+23},
// base = 0 (lanes<16) or 8 (lanes>=16).  Two 16-byte loads.
__device__ __forceinline__ v16h load_tileA(const _Float16* p, int lane) {
  int base = (lane & 16) ? 8 : 0;
  v8h r1 = *(const v8h*)(p + base);
  v8h r2 = *(const v8h*)(p + base + 16);
  v16h a;
#pragma unroll
  for (int i = 0; i < 8; ++i) { a[i] = r1[i]; a[i + 8] = r2[i]; }
  return a;
}
// B 32x16: lane n = column n; K runs {off..off+7} and {off+16..off+23}, off = 0 / 8. Two 16-byte loads.
__device__ __forceinline__ v16h load_tileB(const _Float16* p, int lane) {
  int off = (lane & 16) ? 8 : 0;
  v8h r1 = *(const v8h*)(p + off);
  v8h r2 = *(const v8h*)(p + off + 16);
  v16h b;
#pragma unroll
  for (int i = 0; i < 8; ++i) { b[i] = r1[i]; b[i + 8] = r2[i]; }
  return b;
}

// =====================================================================
// WMMA GEMM on pre-padded f16 operands:
//   C[M,N](f32) = A[M,Kp](f16, lda=Kp-padded) @ W[Npad,Kp](f16)^T + bias[N]
// All operand loads unconditional & 128-bit. One wave per 16x16 tile.
// =====================================================================
__global__ void gemm_wmma_f16_kernel(const _Float16* __restrict__ A, int lda,
                                     const _Float16* __restrict__ W, int ldb,
                                     const float* __restrict__ bias,
                                     float* __restrict__ C, int ldc,
                                     int N, int Kp, int tilesN, int totalTiles)
{
  int wave = blockIdx.x * (blockDim.x >> 5) + (threadIdx.x >> 5);
  if (wave >= totalTiles) return;                  // wave-uniform
  int lane = threadIdx.x & 31;
  int tm = wave / tilesN;
  int tn = wave % tilesN;

  const _Float16* Arow = A + (size_t)(tm * 16 + (lane & 15)) * lda;
  const _Float16* Brow = W + (size_t)(tn * 16 + (lane & 15)) * ldb;

  v8f acc = {};
  for (int k0 = 0; k0 < Kp; k0 += 32) {
    v16h a = load_tileA(Arow + k0, lane);
    v16h b = load_tileB(Brow + k0, lane);
    acc = __builtin_amdgcn_wmma_f32_16x16x32_f16(false, a, false, b,
                                                 (short)0, acc, false, false);
  }

  int n = tn * 16 + (lane & 15);
  if (n < N) {
    float bv = bias ? bias[n] : 0.f;
    int mbase = tm * 16 + ((lane >> 4) << 3);
#pragma unroll
    for (int r = 0; r < 8; ++r)
      C[(size_t)(mbase + r) * ldc + n] = acc[r] + bv;
  }
}

// =====================================================================
// Zero-padded f32 -> f16 conversion (weights): dst[Rp][Cp] from src[R][C]
// =====================================================================
__global__ void cvt_pad_f16_kernel(const float* __restrict__ src, int R, int C,
                                   _Float16* __restrict__ dst, int Rp, int Cp)
{
  int idx = blockIdx.x * blockDim.x + threadIdx.x;
  if (idx >= Rp * Cp) return;
  int r = idx / Cp, c = idx % Cp;
  float v = (r < R && c < C) ? src[r * C + c] : 0.f;
  dst[idx] = (_Float16)v;
}

// =====================================================================
// Embedding gather -> padded f16 [4096][128]
// =====================================================================
__global__ void embed_f16_kernel(const int* __restrict__ wid, const int* __restrict__ uid,
                                 const float* __restrict__ wl, const float* __restrict__ tl,
                                 _Float16* __restrict__ out)
{
  int idx = blockIdx.x * blockDim.x + threadIdx.x;
  if (idx >= ROWS * 128) return;
  int row = idx >> 7, d = idx & 127;
  float v = 0.f;
  if (d < 100)      v = wl[wid[row] * 100 + d];
  else if (d < DH)  v = tl[uid[row] * 25 + (d - 100)];
  out[idx] = (_Float16)v;
}

// =====================================================================
// Persistent LSTM scan: one workgroup (32 waves) per direction.
// Whh LDS-resident f16 [512][128]; per step gates[32,512] = h @ Whh^T via WMMA,
// then cell update. Output written as padded f16 [4096][256] at column offset.
// =====================================================================
__global__ void lstm_scan_kernel(const float* __restrict__ xg_f,
                                 const float* __restrict__ xg_b,
                                 const float* __restrict__ Whh_f,
                                 const float* __restrict__ Whh_b,
                                 _Float16* __restrict__ out)
{
  int dir = blockIdx.x;                 // 0 = forward, 1 = backward
  const float* xg  = dir ? xg_b  : xg_f;
  const float* Whh = dir ? Whh_b : Whh_f;
  int outOff = dir ? DH : 0;

  extern __shared__ char smem[];
  _Float16* sW = (_Float16*)smem;              // [512][128] padded Whh (f16)
  _Float16* sH = sW + 512 * 128;               // [32][128]  h state (f16)
  float*    sC = (float*)(sH + 32 * 128);      // [32][128]  c state (f32)
  float*    sG = sC + 32 * 128;                // [32][512]  gates   (f32)

  int tid = threadIdx.x;

  // preload + zero-pad Whh (f32 [500][125] -> f16 [512][128])
  for (int idx = tid; idx < 512 * 128; idx += blockDim.x) {
    int n = idx >> 7, k = idx & 127;
    float v = (n < G4 && k < DH) ? Whh[n * DH + k] : 0.f;
    sW[idx] = (_Float16)v;
  }
  for (int idx = tid; idx < 32 * 128; idx += blockDim.x) {
    sH[idx] = (_Float16)0.f;
    sC[idx] = 0.f;
  }
  // dir 0 zeroes the pad columns [250,256) of the output once
  if (dir == 0) {
    for (int idx = tid; idx < ROWS * (WELD - H2); idx += blockDim.x) {
      int r = idx / (WELD - H2), c = idx % (WELD - H2);
      out[(size_t)r * WELD + H2 + c] = (_Float16)0.f;
    }
  }
  __syncthreads();

  int w = tid >> 5, lane = tid & 31;

  for (int tt = 0; tt < SEQ; ++tt) {
    int t = dir ? (SEQ - 1 - tt) : tt;

    // ---- gates = h @ Whh^T (+ xg[:,t,:]) -> sG ----
#pragma unroll
    for (int s = 0; s < 2; ++s) {            // two 16-row batch tiles
      v8f acc = {};
      const _Float16* hrow = sH + (s * 16 + (lane & 15)) * 128;
      const _Float16* wrow = sW + (w * 16 + (lane & 15)) * 128;
#pragma unroll
      for (int kc = 0; kc < 4; ++kc) {       // K = 128 = 4 x 32
        v16h a = load_tileA(hrow + kc * 32, lane);
        v16h b = load_tileB(wrow + kc * 32, lane);
        acc = __builtin_amdgcn_wmma_f32_16x16x32_f16(false, a, false, b,
                                                     (short)0, acc, false, false);
      }
      int n = w * 16 + (lane & 15);
      int mbase = s * 16 + ((lane >> 4) << 3);
#pragma unroll
      for (int r = 0; r < 8; ++r) {
        int bb = mbase + r;
        // xg has padded stride 512; pad columns never feed the cell update
        sG[bb * 512 + n] = acc[r] + xg[(size_t)(bb * SEQ + t) * XGLD + n];
      }
    }
    __syncthreads();

    // ---- elementwise LSTM cell (gate order i,f,g,o) ----
    for (int idx = tid; idx < BATCH * DH; idx += blockDim.x) {
      int bb = idx / DH, j = idx % DH;
      float gi = sG[bb * 512 + j];
      float gf = sG[bb * 512 + DH + j];
      float gg = sG[bb * 512 + 2 * DH + j];
      float go = sG[bb * 512 + 3 * DH + j];
      float c = sigf(gf) * sC[bb * 128 + j] + sigf(gi) * tanhf(gg);
      float h = sigf(go) * tanhf(c);
      sC[bb * 128 + j] = c;
      sH[bb * 128 + j] = (_Float16)h;
      out[(size_t)(bb * SEQ + t) * WELD + outOff + j] = (_Float16)h;
    }
    __syncthreads();
  }
}

// =====================================================================
// Biaffine arc scoring + margin + row argmax
// =====================================================================
__global__ void arc_kernel(const float* __restrict__ wh, const float* __restrict__ wm,
                           const float* __restrict__ esw, const float* __restrict__ esb,
                           const int* __restrict__ ta,
                           float* __restrict__ arc_out, float* __restrict__ trees)
{
  int bi = blockIdx.x;            // b*128 + i
  int b = bi >> 7, i = bi & 127;
  int j = threadIdx.x;            // 0..127

  __shared__ float shh[HID];
  __shared__ float sval[SEQ];
  __shared__ int   sidx[SEQ];

  if (j < HID) shh[j] = wh[bi * HID + j];
  __syncthreads();

  const float* wmr = wm + (size_t)(b * SEQ + j) * HID;
  float s = esb[0];
#pragma unroll 4
  for (int h = 0; h < HID; ++h) s += esw[h] * tanhf(shh[h] + wmr[h]);

  int ta0 = (i == 0) ? 0 : ta[bi];
  s += 1.f - ((j == ta0) ? 1.f : 0.f);     // margin augmentation

  arc_out[(size_t)bi * SEQ + j] = s;
  sval[j] = s; sidx[j] = j;
  __syncthreads();

  for (int off = 64; off > 0; off >>= 1) {  // argmax, first-max tie-break
    if (j < off) {
      float v2 = sval[j + off]; int i2 = sidx[j + off];
      if (v2 > sval[j] || (v2 == sval[j] && i2 < sidx[j])) { sval[j] = v2; sidx[j] = i2; }
    }
    __syncthreads();
  }
  if (j == 0) trees[bi] = (float)sidx[0];
}

// gather gold-head rows directly in padded f16 (pad columns copied as zeros)
__global__ void gather_heads_kernel(const _Float16* __restrict__ we,
                                    const int* __restrict__ ta,
                                    _Float16* __restrict__ rhin)
{
  int idx = blockIdx.x * blockDim.x + threadIdx.x;
  if (idx >= ROWS * WELD) return;
  int row = idx >> 8, c = idx & (WELD - 1);
  int b = row >> 7, t = row & 127;
  int head = (t == 0) ? 0 : (ta[row] & 127);
  rhin[idx] = we[(size_t)(b * SEQ + head) * WELD + c];
}

// tanh(rm + rh) -> padded f16 [4096][128]
__global__ void tanh_add_f16_kernel(const float* __restrict__ rm, const float* __restrict__ rh,
                                    _Float16* __restrict__ o)
{
  int idx = blockIdx.x * blockDim.x + threadIdx.x;
  if (idx >= ROWS * 128) return;
  int r = idx >> 7, c = idx & 127;
  float v = (c < HID) ? tanhf(rm[r * HID + c] + rh[r * HID + c]) : 0.f;
  o[idx] = (_Float16)v;
}

__global__ void rel_argmax_kernel(const float* __restrict__ rs, float* __restrict__ pred)
{
  int m = blockIdx.x * blockDim.x + threadIdx.x;
  if (m >= ROWS) return;
  const float* r = rs + (size_t)m * NL;
  float best = r[0]; int bi = 0;
  for (int l = 1; l < NL; ++l) if (r[l] > best) { best = r[l]; bi = l; }
  pred[m] = (float)bi;
}

// =====================================================================
extern "C" void kernel_launch(void* const* d_in, const int* in_sizes, int n_in,
                              void* d_out, int out_size, void* d_ws, size_t ws_size,
                              hipStream_t stream)
{
  const int*   word_ids = (const int*)d_in[0];
  const int*   upos_ids = (const int*)d_in[1];
  const int*   ta       = (const int*)d_in[2];
  const float* wlookup  = (const float*)d_in[3];
  const float* tlookup  = (const float*)d_in[4];
  const float* l0f_Wih = (const float*)d_in[5],  *l0f_Whh = (const float*)d_in[6],  *l0f_b = (const float*)d_in[7];
  const float* l0b_Wih = (const float*)d_in[8],  *l0b_Whh = (const float*)d_in[9],  *l0b_b = (const float*)d_in[10];
  const float* l1f_Wih = (const float*)d_in[11], *l1f_Whh = (const float*)d_in[12], *l1f_b = (const float*)d_in[13];
  const float* l1b_Wih = (const float*)d_in[14], *l1b_Whh = (const float*)d_in[15], *l1b_b = (const float*)d_in[16];
  const float* eh_W = (const float*)d_in[17], *eh_b = (const float*)d_in[18];
  const float* em_W = (const float*)d_in[19], *em_b = (const float*)d_in[20];
  const float* es_W = (const float*)d_in[21], *es_b = (const float*)d_in[22];
  const float* lh_W = (const float*)d_in[23], *lh_b = (const float*)d_in[24];
  const float* lm_W = (const float*)d_in[25], *lm_b = (const float*)d_in[26];
  const float* ls_W = (const float*)d_in[27], *ls_b = (const float*)d_in[28];

  // ---- workspace layout ----
  float* ws = (float*)d_ws;
  float* f_xg0 = ws;                        // [4096][512] f32
  float* f_xg1 = ws + ROWS * XGLD;          // [4096][512] f32
  // reused after LSTMs (inside the xg region):
  float* f_wh = f_xg0;                      // [4096][100] f32
  float* f_wm = f_xg0 + 409600;
  float* f_rh = f_xg0 + 819200;
  float* f_rm = f_xg0 + 1228800;

  _Float16* hb = (_Float16*)(ws + 2 * ROWS * XGLD);
  _Float16* h_words = hb;                   // [4096][128]
  _Float16* h_h0    = hb + 524288;          // [4096][256]
  _Float16* h_we    = hb + 1572864;         // [4096][256]
  _Float16* h_rhin  = hb + 2621440;         // [4096][256]
  _Float16* h_tt    = hb + 3670016;         // [4096][128]
  _Float16* w_l0f   = hb + 4194304;         // [512][128]
  _Float16* w_l0b   = w_l0f + 65536;
  _Float16* w_l1f   = w_l0b + 65536;        // [512][256]
  _Float16* w_l1b   = w_l1f + 131072;
  _Float16* w_eh    = w_l1b + 131072;       // [112][256]
  _Float16* w_em    = w_eh + 28672;
  _Float16* w_lh    = w_em + 28672;
  _Float16* w_lm    = w_lh + 28672;
  _Float16* w_ls    = w_lm + 28672;         // [64][128]

  float* out = (float*)d_out;
  float* o_trees = out;                          // 4096
  float* o_rels  = out + ROWS;                   // 4096
  float* o_arc   = out + 2 * ROWS;               // 524288
  float* o_relsc = out + 2 * ROWS + ROWS * SEQ;  // 204800

  auto cvt = [&](const float* src, int R, int C, _Float16* dst, int Rp, int Cp) {
    int n = Rp * Cp;
    cvt_pad_f16_kernel<<<(n + 255) / 256, 256, 0, stream>>>(src, R, C, dst, Rp, Cp);
  };
  auto gemm = [&](const _Float16* A, int lda, const _Float16* W, int ldb,
                  const float* bias, float* C, int ldc, int N, int Kp) {
    int tilesN = (N + 15) / 16;
    int total = (ROWS / 16) * tilesN;
    gemm_wmma_f16_kernel<<<(total + 7) / 8, 256, 0, stream>>>(
        A, lda, W, ldb, bias, C, ldc, N, Kp, tilesN, total);
  };

  const size_t scan_lds = (512 * 128 + 32 * 128) * sizeof(_Float16)
                        + (32 * 128 + 32 * 512) * sizeof(float);   // 221184 B

  // 0) weight prep: f32 -> zero-padded f16
  cvt(l0f_Wih, G4, DH, w_l0f, 512, 128);
  cvt(l0b_Wih, G4, DH, w_l0b, 512, 128);
  cvt(l1f_Wih, G4, H2, w_l1f, 512, 256);
  cvt(l1b_Wih, G4, H2, w_l1b, 512, 256);
  cvt(eh_W, HID, H2, w_eh, 112, 256);
  cvt(em_W, HID, H2, w_em, 112, 256);
  cvt(lh_W, HID, H2, w_lh, 112, 256);
  cvt(lm_W, HID, H2, w_lm, 112, 256);
  cvt(ls_W, NL, HID, w_ls, 64, 128);

  // 1) embeddings (padded f16)
  embed_f16_kernel<<<(ROWS * 128 + 255) / 256, 256, 0, stream>>>(
      word_ids, upos_ids, wlookup, tlookup, h_words);

  // 2) BiLSTM layer 0
  gemm(h_words, 128, w_l0f, 128, l0f_b, f_xg0, XGLD, G4, 128);
  gemm(h_words, 128, w_l0b, 128, l0b_b, f_xg1, XGLD, G4, 128);
  lstm_scan_kernel<<<2, 1024, scan_lds, stream>>>(f_xg0, f_xg1, l0f_Whh, l0b_Whh, h_h0);

  // 3) BiLSTM layer 1
  gemm(h_h0, WELD, w_l1f, 256, l1f_b, f_xg0, XGLD, G4, 256);
  gemm(h_h0, WELD, w_l1b, 256, l1b_b, f_xg1, XGLD, G4, 256);
  lstm_scan_kernel<<<2, 1024, scan_lds, stream>>>(f_xg0, f_xg1, l1f_Whh, l1b_Whh, h_we);

  // 4) arc scorer projections + biaffine scores + tree argmax
  gemm(h_we, WELD, w_eh, 256, eh_b, f_wh, HID, HID, 256);
  gemm(h_we, WELD, w_em, 256, em_b, f_wm, HID, HID, 256);
  arc_kernel<<<ROWS, SEQ, 0, stream>>>(f_wh, f_wm, es_W, es_b, ta, o_arc, o_trees);

  // 5) relation scorer (gold tree = target_arcs with [:,0]=0)
  gather_heads_kernel<<<(ROWS * WELD + 255) / 256, 256, 0, stream>>>(h_we, ta, h_rhin);
  gemm(h_rhin, WELD, w_lh, 256, lh_b, f_rh, HID, HID, 256);
  gemm(h_we,   WELD, w_lm, 256, lm_b, f_rm, HID, HID, 256);
  tanh_add_f16_kernel<<<(ROWS * 128 + 255) / 256, 256, 0, stream>>>(f_rm, f_rh, h_tt);
  gemm(h_tt, 128, w_ls, 128, ls_b, o_relsc, NL, NL, 128);
  rel_argmax_kernel<<<(ROWS + 255) / 256, 256, 0, stream>>>(o_relsc, o_rels);
}
